// Transformer_12764642803861
// MI455X (gfx1250) — compile-verified, matrix operand order checked by reading, not run on hardware
//
#include <hip/hip_runtime.h>

// ---------------------------------------------------------------------------
// Types for CDNA5 WMMA (wave32, 16x16x32 bf16 -> f32)
// ---------------------------------------------------------------------------
typedef __bf16 bf16;
typedef __attribute__((ext_vector_type(4)))  __bf16 v4bf;
typedef __attribute__((ext_vector_type(8)))  __bf16 v8bf;
typedef __attribute__((ext_vector_type(16))) __bf16 v16bf;
typedef __attribute__((ext_vector_type(8)))  float  v8f;

static __device__ __forceinline__ v8f wmma_bf16(v16bf a, v16bf b, v8f c) {
  return __builtin_amdgcn_wmma_f32_16x16x32_bf16(
      /*neg_a=*/false, a, /*neg_b=*/false, b,
      /*c_mod=*/(short)0, c, /*reuse_a=*/false, /*reuse_b=*/false);
}

// Load one 16x32 bf16 fragment from an LDS row (lane&15 = row/col, lane>>4
// selects the K-half per ISA 7.12.2 16-bit A/B layout).
static __device__ __forceinline__ v16bf frag16x32(const bf16* rowbase, int hi) {
  v8bf lo = *(const v8bf*)(rowbase + hi * 8);
  v8bf hh = *(const v8bf*)(rowbase + 16 + hi * 8);
  v16bf r;
#pragma unroll
  for (int i = 0; i < 8; ++i) { r[i] = lo[i]; r[8 + i] = hh[i]; }
  return r;
}

// CDNA5 async global->LDS copy (ASYNCcnt-tracked, 16B per lane).
static __device__ __forceinline__ void async_copy_b128(unsigned lds_byte_off,
                                                       unsigned long long gaddr) {
  asm volatile("global_load_async_to_lds_b128 %0, %1, off"
               :: "v"(lds_byte_off), "v"(gaddr) : "memory");
}
static __device__ __forceinline__ void wait_async_le8() {
  asm volatile("s_wait_asynccnt 0x8" ::: "memory");
}
static __device__ __forceinline__ void wait_async_le2() {
  asm volatile("s_wait_asynccnt 0x2" ::: "memory");
}
static __device__ __forceinline__ void wait_async_0() {
  asm volatile("s_wait_asynccnt 0x0" ::: "memory");
}

// ---------------------------------------------------------------------------
// Elementwise fp32 -> bf16 convert (4 elems / thread)
// ---------------------------------------------------------------------------
__global__ __launch_bounds__(256) void convert_bf16_kernel(
    const float* __restrict__ in, bf16* __restrict__ out) {
  int i = (blockIdx.x * 256 + threadIdx.x) * 4;
  float4 f = *(const float4*)(in + i);
  v4bf t;
  t[0] = (bf16)f.x; t[1] = (bf16)f.y; t[2] = (bf16)f.z; t[3] = (bf16)f.w;
  *(v4bf*)(out + i) = t;
}

// ---------------------------------------------------------------------------
// Weight convert + transpose: W[K,N] fp32 -> Wt[N,K] bf16 (32x32 LDS tiles)
// ---------------------------------------------------------------------------
__global__ __launch_bounds__(256) void convert_transpose_kernel(
    const float* __restrict__ W, bf16* __restrict__ Wt, int K, int N) {
  __shared__ bf16 tile[32][33];
  const int kt = blockIdx.y * 32, nt = blockIdx.x * 32;
  const int c = threadIdx.x & 31, r0 = threadIdx.x >> 5;
#pragma unroll
  for (int i = 0; i < 4; ++i) {
    int r = r0 + i * 8;
    tile[r][c] = (bf16)W[(size_t)(kt + r) * N + nt + c];
  }
  __syncthreads();
#pragma unroll
  for (int i = 0; i < 4; ++i) {
    int r = r0 + i * 8;
    Wt[(size_t)(nt + r) * K + kt + c] = tile[c][r];
  }
}

// ---------------------------------------------------------------------------
// GEMM: C[M,N] = act(A[M,K]bf16 @ Wt[N,K]^T bf16 + bias).
// Triple-buffered async-LDS pipeline, 64-K tiles, ONE barrier per tile.
// Block tile 128x128 (256 thr, 8 waves), wave tile 32x64.
// ---------------------------------------------------------------------------
template <int N, bool OUTBF, bool RELU>
__global__ __launch_bounds__(256) void gemm_async_kernel(
    const bf16* __restrict__ A, const bf16* __restrict__ Wt,
    const float* __restrict__ bias, void* __restrict__ Cv, int M, int K) {
  __shared__ bf16 sA[3][128 * 64];  // [buf][m][k]   16KB each
  __shared__ bf16 sB[3][128 * 64];  // [buf][n][k]

  const int tid = threadIdx.x;
  const int lane = tid & 31, wave = tid >> 5;
  const int mrow = lane & 15, hi = lane >> 4, n16 = lane & 15;
  const int bm = blockIdx.y * 128, bn = blockIdx.x * 128;
  const int wm = (wave >> 1) * 32, wn = (wave & 1) * 64;

  // Copy mapping: chunk o = tid*16 + j*4096 -> row=(tid>>3)+j*32,
  // kbyte=(tid&7)*16 (128B row of 64 bf16 K).
  unsigned long long gA0 = (unsigned long long)(const char*)A +
                           (size_t)(bm + (tid >> 3)) * (2 * K) + (tid & 7) * 16;
  unsigned long long gB0 = (unsigned long long)(const char*)Wt +
                           (size_t)(bn + (tid >> 3)) * (2 * K) + (tid & 7) * 16;
  const unsigned long long rstep = (unsigned long long)32 * 2 * K;
  const unsigned ldsA = (unsigned)(size_t)(&sA[0][0]) + tid * 16;
  const unsigned ldsB = (unsigned)(size_t)(&sB[0][0]) + tid * 16;

  v8f acc[2][4];
  const v8f vzero = {0.f, 0.f, 0.f, 0.f, 0.f, 0.f, 0.f, 0.f};
#pragma unroll
  for (int mt = 0; mt < 2; ++mt)
#pragma unroll
    for (int nt = 0; nt < 4; ++nt) acc[mt][nt] = vzero;

  auto issue = [&](int buf, unsigned long long koff) {  // 8 asyncs / tile
#pragma unroll
    for (int j = 0; j < 4; ++j) {
      async_copy_b128(ldsA + buf * 16384 + j * 4096, gA0 + j * rstep + koff);
      async_copy_b128(ldsB + buf * 16384 + j * 4096, gB0 + j * rstep + koff);
    }
  };
  auto compute = [&](int buf) {  // 16 WMMA per tile
#pragma unroll
    for (int ks = 0; ks < 64; ks += 32) {
      v16bf bfr[4];
#pragma unroll
      for (int nt = 0; nt < 4; ++nt)
        bfr[nt] = frag16x32(&sB[buf][(wn + nt * 16 + n16) * 64 + ks], hi);
#pragma unroll
      for (int mt = 0; mt < 2; ++mt) {
        v16bf af = frag16x32(&sA[buf][(wm + mt * 16 + mrow) * 64 + ks], hi);
#pragma unroll
        for (int nt = 0; nt < 4; ++nt)
          acc[mt][nt] = wmma_bf16(af, bfr[nt], acc[mt][nt]);
      }
    }
  };

  const int T = K >> 6;  // 64-K tiles
  issue(0, 0);
  if (T > 1) issue(1, 128);
  for (int t = 0; t < T; ++t) {
    // Own tile-t asyncs done (tile t+1 may stay in flight) -> barrier makes
    // it a block-wide guarantee and also proves everyone finished tile t-1.
    if (t + 1 < T) wait_async_le8();
    else           wait_async_0();
    __syncthreads();
    if (t + 2 < T) issue((t + 2) % 3, (unsigned long long)(t + 2) * 128);
    compute(t % 3);
  }

  // Epilogue: C layout -> row = r + 8*hi, col = lane&15.
#pragma unroll
  for (int mt = 0; mt < 2; ++mt) {
#pragma unroll
    for (int nt = 0; nt < 4; ++nt) {
      int col = bn + wn + nt * 16 + n16;
      float bv = bias[col];
#pragma unroll
      for (int r = 0; r < 8; ++r) {
        int row = bm + wm + mt * 16 + r + 8 * hi;
        float vv = acc[mt][nt][r] + bv;
        if (RELU) vv = fmaxf(vv, 0.f);
        if (OUTBF)
          ((bf16*)Cv)[(size_t)row * N + col] = (bf16)vv;
        else
          ((float*)Cv)[(size_t)row * N + col] = vv;
      }
    }
  }
}

// ---------------------------------------------------------------------------
// Flash attention (bf16 in/out): per (b, h, 128-q tile). 4 waves x 32 q rows.
// Triple-buffered K/V tiles, one barrier per tile; online softmax.
// ---------------------------------------------------------------------------
__global__ __launch_bounds__(128) void attn_fa_kernel(
    const bf16* __restrict__ Q, const bf16* __restrict__ Kx,
    const bf16* __restrict__ V, bf16* __restrict__ O) {
  const int S = 1024, LD = 1024;
  __shared__ bf16 sQ[128 * 64];      // [qrow][d]
  __shared__ bf16 sK[3][32 * 64];    // [buf][key][d]
  __shared__ bf16 sVt[3][64 * 32];   // [buf][dv][key]
  __shared__ bf16 sP[4][32 * 32];    // per-wave probs

  const int tid = threadIdx.x;
  const int lane = tid & 31, wave = tid >> 5;
  const int mrow = lane & 15, hi = lane >> 4, n16 = lane & 15;
  const int b = blockIdx.z, h = blockIdx.y;
  const size_t headoff = (size_t)b * S * LD + (size_t)h * 64;
  const bf16* qbase = Q + headoff;
  const bf16* kbase = Kx + headoff;
  const bf16* vbase = V + headoff;
  bf16*       obase = O + headoff;
  const int q0 = blockIdx.x * 128;
  const int wq = wave * 32;

  // Async Q tile: 16KB, 8 chunks/thread.
  {
    unsigned ldsQ = (unsigned)(size_t)sQ + tid * 16;
    unsigned long long gQ = (unsigned long long)(const char*)qbase +
                            (size_t)(q0 + (tid >> 3)) * (LD * 2) + (tid & 7) * 16;
#pragma unroll
    for (int j = 0; j < 8; ++j)
      async_copy_b128(ldsQ + j * 2048, gQ + (size_t)j * 16 * (LD * 2));
  }

  const unsigned ldsK = (unsigned)(size_t)(&sK[0][0]) + tid * 16;
  const unsigned long long gK0 = (unsigned long long)(const char*)kbase +
                                 (size_t)(tid >> 3) * (LD * 2) + (tid & 7) * 16;

  auto issueKV = [&](int buf, int kt) {
    unsigned long long g = gK0 + (unsigned long long)kt * (LD * 2);
    async_copy_b128(ldsK + buf * 4096, g);
    async_copy_b128(ldsK + buf * 4096 + 2048, g + (size_t)16 * (LD * 2));
#pragma unroll
    for (int l = 0; l < 2; ++l) {  // V: manual transposed store [dv][key]
      int c = tid * 2 + l;
      int row = c >> 3, col = (c & 7) * 8;
      v8bf val = *(const v8bf*)(vbase + (size_t)(kt + row) * LD + col);
#pragma unroll
      for (int i = 0; i < 8; ++i) sVt[buf][(col + i) * 32 + row] = val[i];
    }
  };

  float mS[2][8], lS[2][8];
  v8f oacc[2][4];
  const v8f vzero = {0.f, 0.f, 0.f, 0.f, 0.f, 0.f, 0.f, 0.f};
#pragma unroll
  for (int t = 0; t < 2; ++t) {
#pragma unroll
    for (int r = 0; r < 8; ++r) { mS[t][r] = -1e30f; lS[t][r] = 0.f; }
#pragma unroll
    for (int d = 0; d < 4; ++d) oacc[t][d] = vzero;
  }

  const int T = S / 32;  // 32 key tiles
  issueKV(0, 0);
  issueKV(1, 32);
  for (int t = 0; t < T; ++t) {
    if (t + 1 < T) wait_async_le2();  // tile t (and prologue Q) done
    else           wait_async_0();
    __syncthreads();
    if (t + 2 < T) issueKV((t + 2) % 3, (t + 2) * 32);
    const int buf = t % 3;

    // scores[32q x 32k] = Q*K^T
    v8f sacc[2][2];
    sacc[0][0] = vzero; sacc[0][1] = vzero; sacc[1][0] = vzero; sacc[1][1] = vzero;
#pragma unroll
    for (int ks = 0; ks < 64; ks += 32) {
      v16bf bf0 = frag16x32(&sK[buf][(n16)*64 + ks], hi);
      v16bf bf1 = frag16x32(&sK[buf][(n16 + 16) * 64 + ks], hi);
#pragma unroll
      for (int mt = 0; mt < 2; ++mt) {
        v16bf af = frag16x32(&sQ[(wq + mt * 16 + mrow) * 64 + ks], hi);
        sacc[mt][0] = wmma_bf16(af, bf0, sacc[mt][0]);
        sacc[mt][1] = wmma_bf16(af, bf1, sacc[mt][1]);
      }
    }

    // Online softmax; 1/sqrt(DK)=0.125 folded here (fp32 side).
#pragma unroll
    for (int mt = 0; mt < 2; ++mt) {
#pragma unroll
      for (int r = 0; r < 8; ++r) {
        float s0 = sacc[mt][0][r] * 0.125f;
        float s1 = sacc[mt][1][r] * 0.125f;
        float mx = fmaxf(s0, s1);
#pragma unroll
        for (int off = 1; off < 16; off <<= 1)
          mx = fmaxf(mx, __shfl_xor(mx, off, 32));
        float mnew = fmaxf(mS[mt][r], mx);
        float scale = __expf(mS[mt][r] - mnew);
        mS[mt][r] = mnew;
        float p0 = __expf(s0 - mnew);
        float p1 = __expf(s1 - mnew);
        float rsum = p0 + p1;
#pragma unroll
        for (int off = 1; off < 16; off <<= 1)
          rsum += __shfl_xor(rsum, off, 32);
        lS[mt][r] = lS[mt][r] * scale + rsum;
#pragma unroll
        for (int d = 0; d < 4; ++d) oacc[mt][d][r] *= scale;
        int prow = mt * 16 + r + 8 * hi;
        sP[wave][prow * 32 + n16] = (bf16)p0;
        sP[wave][prow * 32 + 16 + n16] = (bf16)p1;
      }
    }

    // ctx += P[32q x 32k] * V[32k x 64dv]  (per-wave sP, no barrier needed)
    v16bf ap0 = frag16x32(&sP[wave][(mrow)*32], hi);
    v16bf ap1 = frag16x32(&sP[wave][(16 + mrow) * 32], hi);
#pragma unroll
    for (int dt = 0; dt < 4; ++dt) {
      v16bf bv = frag16x32(&sVt[buf][(dt * 16 + n16) * 32], hi);
      oacc[0][dt] = wmma_bf16(ap0, bv, oacc[0][dt]);
      oacc[1][dt] = wmma_bf16(ap1, bv, oacc[1][dt]);
    }
  }

  // Normalize, write bf16 context (feeds Wo GEMM directly).
#pragma unroll
  for (int mt = 0; mt < 2; ++mt) {
#pragma unroll
    for (int dt = 0; dt < 4; ++dt) {
#pragma unroll
      for (int r = 0; r < 8; ++r) {
        int row = q0 + wq + mt * 16 + r + 8 * hi;
        int col = dt * 16 + n16;
        obase[(size_t)row * LD + col] = (bf16)(oacc[mt][dt][r] / lS[mt][r]);
      }
    }
  }
}

// ---------------------------------------------------------------------------
// Fused residual-add + LayerNorm; optional extra bf16 output copy.
// ---------------------------------------------------------------------------
__global__ __launch_bounds__(256) void add_ln_kernel(
    const float* __restrict__ A, const float* __restrict__ Bv,
    const float* __restrict__ g, const float* __restrict__ beta,
    float* __restrict__ out, bf16* __restrict__ outb) {
  const int D = 1024;
  const int row = blockIdx.x;
  const float* pa = A + (size_t)row * D;
  const float* pb = Bv + (size_t)row * D;
  const int tid = threadIdx.x, lane = tid & 31, wave = tid >> 5;

  float vals[4];
  float s = 0.f, qq = 0.f;
#pragma unroll
  for (int i = 0; i < 4; ++i) {
    int c = tid + i * 256;
    float vv = pa[c] + pb[c];
    vals[i] = vv;
    s += vv;
    qq += vv * vv;
  }
#pragma unroll
  for (int off = 1; off < 32; off <<= 1) {
    s += __shfl_xor(s, off, 32);
    qq += __shfl_xor(qq, off, 32);
  }
  __shared__ float rs[8], rq[8];
  if (lane == 0) { rs[wave] = s; rq[wave] = qq; }
  __syncthreads();
  s = 0.f; qq = 0.f;
#pragma unroll
  for (int w = 0; w < 8; ++w) { s += rs[w]; qq += rq[w]; }
  float mean = s * (1.f / D);
  float var = qq * (1.f / D) - mean * mean;
  float rstd = rsqrtf(var + 1e-6f);
#pragma unroll
  for (int i = 0; i < 4; ++i) {
    int c = tid + i * 256;
    float o = (vals[i] - mean) * rstd * g[c] + beta[c];
    out[(size_t)row * D + c] = o;
    if (outb) outb[(size_t)row * D + c] = (bf16)o;
  }
}

// ---------------------------------------------------------------------------
// Host orchestration
// ---------------------------------------------------------------------------
extern "C" void kernel_launch(void* const* d_in, const int* in_sizes, int n_in,
                              void* d_out, int out_size, void* d_ws, size_t ws_size,
                              hipStream_t stream) {
  (void)in_sizes; (void)n_in; (void)out_size; (void)ws_size;
  const int L = 3, Bn = 4, S = 1024, D = 1024, H = 16, DFF = 4096;
  const int M = Bn * S;  // 4096

  const float* x0  = (const float*)d_in[0];
  const float* Wq  = (const float*)d_in[1];
  const float* bq  = (const float*)d_in[2];
  const float* Wk  = (const float*)d_in[3];
  const float* bk  = (const float*)d_in[4];
  const float* Wv  = (const float*)d_in[5];
  const float* bv  = (const float*)d_in[6];
  const float* Wo  = (const float*)d_in[7];
  const float* bo  = (const float*)d_in[8];
  const float* W1  = (const float*)d_in[9];
  const float* b1  = (const float*)d_in[10];
  const float* W2  = (const float*)d_in[11];
  const float* b2  = (const float*)d_in[12];
  const float* g1  = (const float*)d_in[13];
  const float* be1 = (const float*)d_in[14];
  const float* g2  = (const float*)d_in[15];
  const float* be2 = (const float*)d_in[16];

  const size_t NM = (size_t)M * D;  // 4M elems
  char* p = (char*)d_ws;
  float* att  = (float*)p; p += NM * 4;
  float* out1 = (float*)p; p += NM * 4;
  float* ffn  = (float*)p; p += NM * 4;
  float* xb32 = (float*)p; p += NM * 4;
  bf16* xb16  = (bf16*)p; p += NM * 2;
  bf16* qb    = (bf16*)p; p += NM * 2;
  bf16* kb    = (bf16*)p; p += NM * 2;
  bf16* vb    = (bf16*)p; p += NM * 2;
  bf16* ctxb  = (bf16*)p; p += NM * 2;
  bf16* out1b = (bf16*)p; p += NM * 2;
  bf16* h1b   = (bf16*)p; p += (size_t)M * DFF * 2;
  bf16* Wqt = (bf16*)p; p += (size_t)D * D * 2;
  bf16* Wkt = (bf16*)p; p += (size_t)D * D * 2;
  bf16* Wvt = (bf16*)p; p += (size_t)D * D * 2;
  bf16* Wot = (bf16*)p; p += (size_t)D * D * 2;
  bf16* W1t = (bf16*)p; p += (size_t)D * DFF * 2;
  bf16* W2t = (bf16*)p; p += (size_t)DFF * D * 2;

  dim3 blk(256);
  dim3 gProj(D / 128, M / 128);   // (8, 32)
  dim3 gFF1(DFF / 128, M / 128);  // (32, 32)
  dim3 gAttn(S / 128, H, Bn);
  dim3 gT_DD(D / 32, D / 32);
  dim3 gT_DF(DFF / 32, D / 32);
  dim3 gT_FD(D / 32, DFF / 32);

  convert_bf16_kernel<<<NM / 1024, blk, 0, stream>>>(x0, xb16);

  const float* x32 = x0;
  for (int l = 0; l < L; ++l) {
    const size_t wOfs = (size_t)l * D * D;
    const size_t fOfs = (size_t)l * D * DFF;
    const size_t vOfs = (size_t)l * D;
    const size_t fvOfs = (size_t)l * DFF;

    convert_transpose_kernel<<<gT_DD, blk, 0, stream>>>(Wq + wOfs, Wqt, D, D);
    convert_transpose_kernel<<<gT_DD, blk, 0, stream>>>(Wk + wOfs, Wkt, D, D);
    convert_transpose_kernel<<<gT_DD, blk, 0, stream>>>(Wv + wOfs, Wvt, D, D);
    convert_transpose_kernel<<<gT_DD, blk, 0, stream>>>(Wo + wOfs, Wot, D, D);
    convert_transpose_kernel<<<gT_DF, blk, 0, stream>>>(W1 + fOfs, W1t, D, DFF);
    convert_transpose_kernel<<<gT_FD, blk, 0, stream>>>(W2 + fOfs, W2t, DFF, D);

    gemm_async_kernel<1024, true, false><<<gProj, blk, 0, stream>>>(xb16, Wqt, bq + vOfs, qb, M, D);
    gemm_async_kernel<1024, true, false><<<gProj, blk, 0, stream>>>(xb16, Wkt, bk + vOfs, kb, M, D);
    gemm_async_kernel<1024, true, false><<<gProj, blk, 0, stream>>>(xb16, Wvt, bv + vOfs, vb, M, D);

    attn_fa_kernel<<<gAttn, dim3(128), 0, stream>>>(qb, kb, vb, ctxb);

    gemm_async_kernel<1024, false, false><<<gProj, blk, 0, stream>>>(ctxb, Wot, bo + vOfs, att, M, D);
    add_ln_kernel<<<M, blk, 0, stream>>>(x32, att, g1 + vOfs, be1 + vOfs, out1, out1b);

    gemm_async_kernel<4096, true, true><<<gFF1, blk, 0, stream>>>(out1b, W1t, b1 + fvOfs, h1b, M, D);
    gemm_async_kernel<1024, false, false><<<gProj, blk, 0, stream>>>(h1b, W2t, b2 + vOfs, ffn, M, DFF);

    float* xo32 = (l == L - 1) ? (float*)d_out : xb32;
    bf16* xo16 = (l == L - 1) ? nullptr : xb16;
    add_ln_kernel<<<M, blk, 0, stream>>>(out1, ffn, g2 + vOfs, be2 + vOfs, xo32, xo16);
    x32 = xb32;
  }
}
